// FeatureResponseGenerator_15461882266358
// MI455X (gfx1250) — matrix operand order repaired and checked by reading, not compiled, not run on hardware
//
#include <hip/hip_runtime.h>

typedef __attribute__((ext_vector_type(16))) _Float16 v16h;
typedef __attribute__((ext_vector_type(8)))  _Float16 h8;
typedef __attribute__((ext_vector_type(8)))  float    v8f;

#define B_   4
#define C_   128
#define H_   128
#define W_   160
#define HW_  (H_ * W_)      // 20480
#define S_   512

#define S_BLK 64            // s rows per block
#define N_BLK 256           // hw cols per block
#define KC    32            // c chunk per LDS stage (one WMMA K)
#define A_STRIDE 136        // 128 c + 8 pad (halves)
#define B_STRIDE 40         // 32 c + 8 pad (halves)

#define SAMPLED_HALFS  (B_ * C_ * S_)        // 262144 halves
#define SUMS_OFF_BYTES (SAMPLED_HALFS * 2)   // 524288 B

// ---------------------------------------------------------------- zero sums
__global__ void frg_zero(float* __restrict__ sums) {
    int tid = blockIdx.x * blockDim.x + threadIdx.x;
    if (tid < B_ * S_) sums[tid] = 0.0f;
}

// ------------------------------------------------- gather sampled -> f16 ws
// sampled[b][c][s] = (f16) src[b][c][loc[b][s]]
__global__ void frg_gather(const float* __restrict__ src,
                           const int*   __restrict__ locs,
                           _Float16*    __restrict__ sampled) {
    int tid = blockIdx.x * blockDim.x + threadIdx.x;   // < B_*C_*S_
    int b   = tid >> 16;          // C_*S_ == 65536
    int rem = tid & 65535;
    int c   = rem >> 9;           // 0..127
    int s   = rem & 511;          // 0..511
    int loc = locs[b * S_ + s];
    sampled[tid] = (_Float16)src[((size_t)(b * C_ + c)) * HW_ + loc];
}

// ----------------------------------------------------------- main WMMA GEMM
// mode 0: accumulate per-(b,s) row sums of exp(10*corr - 8) into `sums`
// mode 1: write exp(10*corr - 8) / sums[b][s] to out
__global__ __launch_bounds__(256)
void frg_main(const float*    __restrict__ tgt,
              const _Float16* __restrict__ sampled,
              float*          __restrict__ sums,
              float*          __restrict__ out,
              int mode) {
    __shared__ _Float16 Asm[S_BLK * A_STRIDE];   // [s][c]   17.4 KB
    __shared__ _Float16 Bsm[N_BLK * B_STRIDE];   // [hw][c]  20.5 KB

    const int tid  = threadIdx.x;
    const int lane = tid & 31;
    const int wave = tid >> 5;
    const int hf   = lane >> 4;   // 0: lanes 0-15, 1: lanes 16-31
    const int l16  = lane & 15;

    const int hw0 = blockIdx.x * N_BLK;
    const int s0  = blockIdx.y * S_BLK;
    const int b   = blockIdx.z;

    const int s_wave  = (wave >> 1) * 16;   // 0/16/32/48
    const int hw_wave = (wave & 1) * 128;   // 0/128

    // ---- stage A (whole 128 x 64 tile, transposed to [s][c]) once
    {
        const _Float16* base = sampled + (size_t)b * C_ * S_ + s0;
#pragma unroll
        for (int i = 0; i < 4; ++i) {
            int slot = tid + i * 256;        // 0..1023
            int c    = slot >> 3;            // 0..127
            int sg   = slot & 7;             // group of 8 s
            h8 v = *(const h8*)(base + (size_t)c * S_ + sg * 8);
#pragma unroll
            for (int j = 0; j < 8; ++j)
                Asm[(sg * 8 + j) * A_STRIDE + c] = v[j];
        }
    }

    v8f acc[8] = {};   // 8 tiles of 16(s) x 16(hw), f32

    for (int ck = 0; ck < C_; ck += KC) {
        __syncthreads();   // A visible (iter 0) / prev-iter frag reads done
        // ---- stage B chunk: tgt[b][ck..ck+31][hw0..hw0+255] -> Bsm[hw][c]
        const float* tb = tgt + ((size_t)b * C_ + ck) * HW_ + hw0;
#pragma unroll
        for (int i = 0; i < 8; ++i) {
            int slot = tid + i * 256;        // 2048 float4 slots
            int r    = slot >> 6;            // c row 0..31
            int c4   = slot & 63;            // float4 along hw
            float4 f = *(const float4*)(tb + (size_t)r * HW_ + c4 * 4);
            int col  = c4 * 4;
            Bsm[(col + 0) * B_STRIDE + r] = (_Float16)f.x;
            Bsm[(col + 1) * B_STRIDE + r] = (_Float16)f.y;
            Bsm[(col + 2) * B_STRIDE + r] = (_Float16)f.z;
            Bsm[(col + 3) * B_STRIDE + r] = (_Float16)f.w;
        }
        __syncthreads();

        // ---- A fragment (ISA 16-bit A 16x32 layout): k = idx + 8*hf (+8 if idx>=8)
        const _Float16* arow = &Asm[(s_wave + l16) * A_STRIDE + ck + 8 * hf];
        h8 alo = *(const h8*)(arow);        // k = 8*hf + 0..7
        h8 ahi = *(const h8*)(arow + 16);   // k = 8*hf + 16..23
        v16h afrag = __builtin_shufflevector(alo, ahi,
            0, 1, 2, 3, 4, 5, 6, 7, 8, 9, 10, 11, 12, 13, 14, 15);

#pragma unroll
        for (int t = 0; t < 8; ++t) {
            // B fragment (32x16 B layout): k = idx + 8*hf (+8 if idx>=8)
            const _Float16* brow = &Bsm[(hw_wave + t * 16 + l16) * B_STRIDE + 8 * hf];
            h8 blo = *(const h8*)(brow);
            h8 bhi = *(const h8*)(brow + 16);
            v16h bfrag = __builtin_shufflevector(blo, bhi,
                0, 1, 2, 3, 4, 5, 6, 7, 8, 9, 10, 11, 12, 13, 14, 15);
            acc[t] = __builtin_amdgcn_wmma_f32_16x16x32_f16(
                /*neg_a=*/false, afrag, /*neg_b=*/false, bfrag,
                /*c_mod=*/(short)0, acc[t], /*reuse_a=*/false, /*reuse_b=*/false);
        }
    }

    // m = exp(SCALE*((0.5*corr+0.5)-THRESHOLD)) = exp(10*corr - 8)
    if (mode == 0) {
        float psum[8];
#pragma unroll
        for (int r = 0; r < 8; ++r) psum[r] = 0.0f;
#pragma unroll
        for (int t = 0; t < 8; ++t)
#pragma unroll
            for (int r = 0; r < 8; ++r)
                psum[r] += __expf(fmaf(10.0f, acc[t][r], -8.0f));
        // rows: lanes 0-15 hold M=r, lanes 16-31 hold M=8+r; sum over 16 n-lanes
#pragma unroll
        for (int r = 0; r < 8; ++r) {
            float v = psum[r];
            v += __shfl_xor(v, 1);
            v += __shfl_xor(v, 2);
            v += __shfl_xor(v, 4);
            v += __shfl_xor(v, 8);
            if (l16 == 0)
                atomicAdd(&sums[b * S_ + s0 + s_wave + hf * 8 + r], v);
        }
    } else {
#pragma unroll
        for (int r = 0; r < 8; ++r) {
            int s = s0 + s_wave + hf * 8 + r;
            float inv = 1.0f / sums[b * S_ + s];
            float* orow = out + ((size_t)(b * S_ + s)) * HW_ + hw0 + hw_wave + l16;
#pragma unroll
            for (int t = 0; t < 8; ++t)
                orow[t * 16] = __expf(fmaf(10.0f, acc[t][r], -8.0f)) * inv;
        }
    }
}

// ---------------------------------------------------------------------------
extern "C" void kernel_launch(void* const* d_in, const int* in_sizes, int n_in,
                              void* d_out, int out_size, void* d_ws, size_t ws_size,
                              hipStream_t stream) {
    const float* src  = (const float*)d_in[0];   // [B,C,H,W] f32
    const float* tgt  = (const float*)d_in[1];   // [B,C,H,W] f32
    const int*   locs = (const int*)d_in[2];     // [B,S,1] int
    // d_in[3] (boundaries) is unused by the reference computation.

    float*     out     = (float*)d_out;                          // [B,S,H,W] f32
    _Float16*  sampled = (_Float16*)d_ws;                        // 512 KB f16
    float*     sums    = (float*)((char*)d_ws + SUMS_OFF_BYTES); // 8 KB f32

    frg_zero<<<(B_ * S_ + 255) / 256, 256, 0, stream>>>(sums);
    frg_gather<<<(B_ * C_ * S_) / 256, 256, 0, stream>>>(src, locs, sampled);

    dim3 grid(HW_ / N_BLK, S_ / S_BLK, B_);   // 80 x 8 x 4 = 2560 blocks
    frg_main<<<grid, 256, 0, stream>>>(tgt, sampled, sums, out, 0);
    frg_main<<<grid, 256, 0, stream>>>(tgt, sampled, sums, out, 1);
}
